// EnhancedLSTMAttentionRecommender_36575941493198
// MI455X (gfx1250) — compile-verified, matrix operand order checked by reading, not run on hardware
//
#include <hip/hip_runtime.h>
#include <hip/hip_bf16.h>

// ---------------- problem constants ----------------
#define kB    256
#define kS    100
#define kK    20
#define kEMB  128
#define kCEMB 32
#define kIN0  160     // EMB + CAT_EMB
#define kH    256
#define kG    1024    // 4*H
#define kNP   50001   // num_products + 1
#define kF1   128     // HID/2

typedef unsigned short u16;
typedef __bf16 bf16_t;
typedef __attribute__((ext_vector_type(16))) bf16_t v16bf;
typedef __attribute__((ext_vector_type(8)))  bf16_t v8bf;
typedef __attribute__((ext_vector_type(8)))  float  v8f;
typedef int v4i __attribute__((ext_vector_type(4)));
typedef __attribute__((address_space(1))) v4i* gv4i_p;   // global int4*
typedef __attribute__((address_space(3))) v4i* lv4i_p;   // LDS int4*

// ---- CDNA5 async-to-LDS path (guarded so compile never breaks) ----
#if defined(__has_builtin)
#  if __has_builtin(__builtin_amdgcn_global_load_async_to_lds_b128)
#    define USE_ASYNC_LDS 1
#  endif
#endif
#ifndef USE_ASYNC_LDS
#  define USE_ASYNC_LDS 0
#endif

#if defined(__has_builtin) && __has_builtin(__builtin_amdgcn_s_wait_asynccnt)
#  define WAIT_ASYNCCNT0() __builtin_amdgcn_s_wait_asynccnt(0)
#else
#  define WAIT_ASYNCCNT0() asm volatile("s_wait_asynccnt 0" ::: "memory")
#endif

// ---------------- small helpers ----------------
__device__ __forceinline__ u16 f2bf(float f) {
  unsigned u = __float_as_uint(f);
  unsigned r = u + 0x7fffu + ((u >> 16) & 1u);   // round-to-nearest-even
  return (u16)(r >> 16);
}
__device__ __forceinline__ float bf2f(u16 h) {
  return __uint_as_float(((unsigned)h) << 16);
}
__device__ __forceinline__ float sigm(float x) {
  return 1.0f / (1.0f + __expf(-x));
}

// Load a 16x32 bf16 A-fragment (row-major source, ld elements per row).
// CDNA5 layout: lanes 0-15 hold row M=lane, K = {0..7, 16..23};
// lanes 16-31 hold row M=lane-16, K = {8..15, 24..31}.
__device__ __forceinline__ v16bf load_frag(const u16* __restrict__ base, int ld,
                                           int r0, int k0, int lane) {
  int half = lane >> 4;
  int r    = lane & 15;
  const u16* p = base + (long)(r0 + r) * ld + k0 + half * 8;
  v8bf lo = *(const v8bf*)p;
  v8bf hi = *(const v8bf*)(p + 16);
  v16bf f;
#pragma unroll
  for (int i = 0; i < 8; ++i) { f[i] = lo[i]; f[i + 8] = hi[i]; }
  return f;
}

#define BSTRIDE 40   // padded LDS row stride (u16) -> conflict-free ds_load_b128

// B fragment from the LDS-staged 64x32 panel; jt selects the 16-col subtile.
__device__ __forceinline__ v16bf load_fragB_lds(const u16* Bt, int jt, int lane) {
  int half = lane >> 4;
  int r    = lane & 15;
  const u16* p = Bt + (jt * 16 + r) * BSTRIDE + half * 8;
  v8bf lo = *(const v8bf*)p;
  v8bf hi = *(const v8bf*)(p + 16);
  v16bf f;
#pragma unroll
  for (int i = 0; i < 8; ++i) { f[i] = lo[i]; f[i + 8] = hi[i]; }
  return f;
}

// ---------------- generic bf16 WMMA GEMM ----------------
// out[M,N] (row stride ldo) = A[M,K] * Bw[N,K]^T (+bias0 +bias1) (+out if accumulate)
// Block: 256 threads = 8 waves -> 128x64 output tile; wave w -> rows w*16..w*16+15.
// B panel (64 cols x 32 K) staged in LDS per k-step; one A fragment drives 4 WMMAs.
__global__ void __launch_bounds__(256)
gemm_bf16_wmma(const u16* __restrict__ A, int lda,
               const u16* __restrict__ Bw, int ldb,
               const float* __restrict__ bias0,
               const float* __restrict__ bias1,
               float* __restrict__ out, long ldo,
               int M, int N, int K, int accumulate) {
  __shared__ u16 Bt[64 * BSTRIDE];
  int tid  = threadIdx.x;
  int wave = tid >> 5;
  int lane = tid & 31;
  int nb64 = (N + 63) >> 6;
  int bm = blockIdx.x / nb64;
  int bn = blockIdx.x % nb64;
  int row0    = bm * 128 + wave * 16;
  int colbase = bn * 64;

  v8f acc[4] = {};

  // per-thread staging slot: 64 rows x 4 chunks of 16B
  int srow   = tid >> 2;       // 0..63
  int schunk = tid & 3;        // 0..3
  int gbrow  = colbase + srow;
  if (gbrow >= N) gbrow = N - 1;
  const u16* gB = Bw + (long)gbrow * ldb + schunk * 8;
  u16*       lB = Bt + srow * BSTRIDE + schunk * 8;

  for (int k0 = 0; k0 < K; k0 += 32) {
#if USE_ASYNC_LDS
    __builtin_amdgcn_global_load_async_to_lds_b128(
        (gv4i_p)(gB + k0), (lv4i_p)lB, 0, 0);
    WAIT_ASYNCCNT0();
    __syncthreads();
#else
    { uint4 tmpv = *(const uint4*)(gB + k0); *(uint4*)lB = tmpv; }
    __syncthreads();
#endif
    if (row0 < M) {
      if (k0 + 32 < K)
        __builtin_prefetch(A + (long)(row0 + (lane & 15)) * lda + k0 + 32, 0, 3);
      v16bf a = load_frag(A, lda, row0, k0, lane);
      v16bf bf[4];
#pragma unroll
      for (int j = 0; j < 4; ++j) bf[j] = load_fragB_lds(Bt, j, lane);
#pragma unroll
      for (int j = 0; j < 4; ++j) {
        acc[j] = __builtin_amdgcn_wmma_f32_16x16x32_bf16(
            false, a, false, bf[j], (short)0, acc[j], false, false);
      }
    }
    __syncthreads();
  }

  if (row0 >= M) return;
  // C/D layout: lanes 0-15 -> M = 0..7 in VGPR 0..7; lanes 16-31 -> M = 8..15.
  int n  = lane & 15;
  int mb = (lane >> 4) * 8;
#pragma unroll
  for (int j = 0; j < 4; ++j) {
    int col = colbase + j * 16 + n;
    if (col < N) {
      float badd = (bias0 ? bias0[col] : 0.0f) + (bias1 ? bias1[col] : 0.0f);
#pragma unroll
      for (int v = 0; v < 8; ++v) {
        long idx  = (long)(row0 + mb + v) * ldo + col;
        float val = acc[j][v] + badd;
        if (accumulate) val += out[idx];
        out[idx] = val;
      }
    }
  }
}

// ---------------- f32 -> bf16 conversion ----------------
__global__ void f32_to_bf16_kernel(const float* __restrict__ in,
                                   u16* __restrict__ out, long n) {
  long i = (long)blockIdx.x * blockDim.x + threadIdx.x;
  if (i < n) out[i] = f2bf(in[i]);
}

// ---------------- embedding + mean-pool + category ----------------
__global__ void embed_kernel(const int* __restrict__ seq,      // [B,S,K]
                             const int* __restrict__ p2c,      // [NP]
                             const float* __restrict__ pemb,   // [NP,EMB]
                             const float* __restrict__ cemb,   // [NC,CEMB]
                             u16* __restrict__ combined) {     // [S,B,IN0]
  int bs = blockIdx.x;            // = b*S + s
  int b = bs / kS, s = bs % kS;
  int t = threadIdx.x;            // 160 threads
  const int* basket = seq + ((long)b * kS + s) * kK;
  __shared__ int firstp;
  if (t == 0) {
    int fp = 0;
#pragma unroll
    for (int k = 0; k < kK; ++k) {
      int p = basket[k];
      if (p != 0) { fp = p; break; }
    }
    firstp = fp;
  }
  float val = 0.0f;
  if (t < kEMB) {
    float acc = 0.0f;
#pragma unroll 4
    for (int k = 0; k < kK; ++k) acc += pemb[(long)basket[k] * kEMB + t];
    val = acc * (1.0f / kK);
  }
  __syncthreads();
  if (t >= kEMB) {
    int cid = p2c[firstp];
    val = cemb[(long)cid * kCEMB + (t - kEMB)];
  }
  combined[((long)s * kB + b) * kIN0 + t] = f2bf(val);
}

// ---------------- LSTM state init ----------------
__global__ void init_state_kernel(float* __restrict__ c, u16* __restrict__ h, long n) {
  long i = (long)blockIdx.x * blockDim.x + threadIdx.x;
  if (i < n) { c[i] = 0.0f; h[i] = 0; }
}

// ---------------- LSTM pointwise gate update ----------------
__global__ void lstm_point_kernel(const float* __restrict__ g,  // [B,4H]
                                  float* __restrict__ c,        // [B,H]
                                  u16* __restrict__ h_prev,     // [B,H] bf16
                                  u16* __restrict__ h_all) {    // [B,H] slice at t
  int tid = blockIdx.x * blockDim.x + threadIdx.x;
  if (tid >= kB * kH) return;
  int b = tid / kH, j = tid % kH;
  const float* gr = g + (long)b * kG;
  float ig = sigm(gr[j]);
  float fg = sigm(gr[kH + j]);
  float gg = tanhf(gr[2 * kH + j]);
  float og = sigm(gr[3 * kH + j]);
  float cn = fg * c[tid] + ig * gg;
  c[tid] = cn;
  float h = og * tanhf(cn);
  u16 hb = f2bf(h);
  h_prev[tid] = hb;
  h_all[tid]  = hb;
}

// ---------------- attention pooling ----------------
__global__ void attn_kernel(const u16* __restrict__ h1,        // [S,B,H] bf16
                            const float* __restrict__ attn_w,  // [H]
                            const float* __restrict__ attn_b,  // [1]
                            float* __restrict__ attw_out,      // [B,S]
                            u16* __restrict__ ctx_bf) {        // [B,H]
  int b = blockIdx.x;
  int t = threadIdx.x;   // 256 threads
  __shared__ float sc[kS];
  __shared__ float w[kS];
  __shared__ float inv_s;
  if (t < kS) {
    const u16* hr = h1 + ((long)t * kB + b) * kH;
    float acc = 0.0f;
    for (int j = 0; j < kH; ++j) acc += bf2f(hr[j]) * attn_w[j];
    sc[t] = acc + attn_b[0];
  }
  __syncthreads();
  if (t == 0) {
    float m = sc[0];
    for (int s = 1; s < kS; ++s) m = fmaxf(m, sc[s]);
    float sum = 0.0f;
    for (int s = 0; s < kS; ++s) { float e = __expf(sc[s] - m); w[s] = e; sum += e; }
    inv_s = 1.0f / sum;
  }
  __syncthreads();
  float inv = inv_s;
  if (t < kS) attw_out[(long)b * kS + t] = w[t] * inv;
  float acc = 0.0f;
  for (int s = 0; s < kS; ++s)
    acc += (w[s] * inv) * bf2f(h1[((long)s * kB + b) * kH + t]);
  ctx_bf[(long)b * kH + t] = f2bf(acc);
}

// ---------------- relu + convert to bf16 ----------------
__global__ void relu_bf16_kernel(const float* __restrict__ in,
                                 u16* __restrict__ out, long n) {
  long i = (long)blockIdx.x * blockDim.x + threadIdx.x;
  if (i < n) out[i] = f2bf(fmaxf(in[i], 0.0f));
}

// ---------------- host-side launchers ----------------
static inline void launch_gemm(const u16* A, int lda, const u16* Bw, int ldb,
                               const float* bias0, const float* bias1,
                               float* out, long ldo, int M, int N, int K,
                               int accumulate, hipStream_t stream) {
  long mb = (M + 127) / 128;
  long nb = (N + 63) / 64;
  long blocks = mb * nb;
  gemm_bf16_wmma<<<dim3((unsigned)blocks), dim3(256), 0, stream>>>(
      A, lda, Bw, ldb, bias0, bias1, out, ldo, M, N, K, accumulate);
}

static inline void launch_cvt(const float* in, u16* out, long n, hipStream_t stream) {
  long blocks = (n + 255) / 256;
  f32_to_bf16_kernel<<<dim3((unsigned)blocks), dim3(256), 0, stream>>>(in, out, n);
}

extern "C" void kernel_launch(void* const* d_in, const int* in_sizes, int n_in,
                              void* d_out, int out_size, void* d_ws, size_t ws_size,
                              hipStream_t stream) {
  // ---- inputs (setup_inputs order) ----
  const int*   seq    = (const int*)  d_in[0];
  const int*   p2c    = (const int*)  d_in[1];
  const float* pemb   = (const float*)d_in[2];
  const float* cemb   = (const float*)d_in[3];
  const float* W_ih0  = (const float*)d_in[4];
  const float* W_hh0  = (const float*)d_in[5];
  const float* b_ih0  = (const float*)d_in[6];
  const float* b_hh0  = (const float*)d_in[7];
  const float* W_ih1  = (const float*)d_in[8];
  const float* W_hh1  = (const float*)d_in[9];
  const float* b_ih1  = (const float*)d_in[10];
  const float* b_hh1  = (const float*)d_in[11];
  const float* attn_w = (const float*)d_in[12];
  const float* attn_b = (const float*)d_in[13];
  const float* fc1_w  = (const float*)d_in[14];
  const float* fc1_b  = (const float*)d_in[15];
  const float* fc2_w  = (const float*)d_in[16];
  const float* fc2_b  = (const float*)d_in[17];

  float* out_logits = (float*)d_out;                 // [B,NP]
  float* out_attw   = out_logits + (size_t)kB * kNP; // [B,S]

  // ---- workspace carve-up (256B aligned) ----
  char* w = (char*)d_ws;
  size_t off = 0;
  auto alloc = [&](size_t bytes) {
    void* p = w + off;
    off += (bytes + 255) & ~(size_t)255;
    return p;
  };
  u16*   combined = (u16*)  alloc((size_t)kS * kB * kIN0 * 2);   // [S,B,IN0]
  u16*   Wih0b    = (u16*)  alloc((size_t)kG * kIN0 * 2);
  u16*   Whh0b    = (u16*)  alloc((size_t)kG * kH * 2);
  u16*   Wih1b    = (u16*)  alloc((size_t)kG * kH * 2);
  u16*   Whh1b    = (u16*)  alloc((size_t)kG * kH * 2);
  u16*   fc1wb    = (u16*)  alloc((size_t)kF1 * kH * 2);
  u16*   fc2wb    = (u16*)  alloc((size_t)kNP * kF1 * 2);
  float* xg       = (float*)alloc((size_t)kS * kB * kG * 4);     // [S,B,4H]
  u16*   h_prev   = (u16*)  alloc((size_t)kB * kH * 2);
  float* c_state  = (float*)alloc((size_t)kB * kH * 4);
  u16*   h0_all   = (u16*)  alloc((size_t)kS * kB * kH * 2);     // [S,B,H]
  u16*   h1_all   = (u16*)  alloc((size_t)kS * kB * kH * 2);     // [S,B,H]
  u16*   ctx_bf   = (u16*)  alloc((size_t)kB * kH * 2);
  float* fc1_out  = (float*)alloc((size_t)kB * kF1 * 4);
  u16*   x_bf     = (u16*)  alloc((size_t)kB * kF1 * 2);
  (void)ws_size; (void)in_sizes; (void)n_in; (void)out_size;

  // ---- 1) convert weights to bf16 ----
  launch_cvt(W_ih0, Wih0b, (long)kG * kIN0, stream);
  launch_cvt(W_hh0, Whh0b, (long)kG * kH,   stream);
  launch_cvt(W_ih1, Wih1b, (long)kG * kH,   stream);
  launch_cvt(W_hh1, Whh1b, (long)kG * kH,   stream);
  launch_cvt(fc1_w, fc1wb, (long)kF1 * kH,  stream);
  launch_cvt(fc2_w, fc2wb, (long)kNP * kF1, stream);

  // ---- 2) embeddings -> combined [S,B,IN0] bf16 ----
  embed_kernel<<<dim3(kB * kS), dim3(kIN0), 0, stream>>>(seq, p2c, pemb, cemb, combined);

  // ---- 3) layer 0 ----
  launch_gemm(combined, kIN0, Wih0b, kIN0, b_ih0, b_hh0,
              xg, kG, kS * kB, kG, kIN0, /*acc=*/0, stream);
  init_state_kernel<<<dim3((kB * kH + 255) / 256), dim3(256), 0, stream>>>(
      c_state, h_prev, (long)kB * kH);
  for (int t = 0; t < kS; ++t) {
    float* g_t = xg + (long)t * kB * kG;
    launch_gemm(h_prev, kH, Whh0b, kH, nullptr, nullptr,
                g_t, kG, kB, kG, kH, /*acc=*/1, stream);
    lstm_point_kernel<<<dim3((kB * kH + 255) / 256), dim3(256), 0, stream>>>(
        g_t, c_state, h_prev, h0_all + (long)t * kB * kH);
  }

  // ---- 4) layer 1 ----
  launch_gemm(h0_all, kH, Wih1b, kH, b_ih1, b_hh1,
              xg, kG, kS * kB, kG, kH, /*acc=*/0, stream);
  init_state_kernel<<<dim3((kB * kH + 255) / 256), dim3(256), 0, stream>>>(
      c_state, h_prev, (long)kB * kH);
  for (int t = 0; t < kS; ++t) {
    float* g_t = xg + (long)t * kB * kG;
    launch_gemm(h_prev, kH, Whh1b, kH, nullptr, nullptr,
                g_t, kG, kB, kG, kH, /*acc=*/1, stream);
    lstm_point_kernel<<<dim3((kB * kH + 255) / 256), dim3(256), 0, stream>>>(
        g_t, c_state, h_prev, h1_all + (long)t * kB * kH);
  }

  // ---- 5) attention ----
  attn_kernel<<<dim3(kB), dim3(kH), 0, stream>>>(h1_all, attn_w, attn_b,
                                                 out_attw, ctx_bf);

  // ---- 6) fc1 + relu ----
  launch_gemm(ctx_bf, kH, fc1wb, kH, fc1_b, nullptr,
              fc1_out, kF1, kB, kF1, kH, /*acc=*/0, stream);
  relu_bf16_kernel<<<dim3((kB * kF1 + 255) / 256), dim3(256), 0, stream>>>(
      fc1_out, x_bf, (long)kB * kF1);

  // ---- 7) fc2 -> logits ----
  launch_gemm(x_bf, kF1, fc2wb, kF1, fc2_b, nullptr,
              out_logits, kNP, kB, kNP, kF1, /*acc=*/0, stream);
}
